// MyInfoNCELossNoSeg_6597069766920
// MI455X (gfx1250) — matrix operand order repaired and checked by reading, not compiled, not run on hardware
//
#include <hip/hip_runtime.h>
#include <hip/hip_bf16.h>

#define BS   4
#define NPTS 8192
#define FDIM 512
#define SM   2048
#define BM   128
#define BN   128
#define BK   32
#define NKT  (FDIM / BK) // 16 K-steps
#define NCB  (SM / BN)   // 16 column blocks per row

typedef __attribute__((ext_vector_type(16))) __bf16 v16bf;
typedef __attribute__((ext_vector_type(8)))  float  v8f;

union V16Frag { v16bf bf; uint4 q[2]; };

// exp(x / 0.1) == exp2(x * 10 / ln2)
static constexpr float K_EXP2 = 14.426950408889634f;

__device__ __forceinline__ unsigned short f32_to_bf16_rne(float f) {
    unsigned int u = __float_as_uint(f);
    unsigned int lsb = (u >> 16) & 1u;
    u += 0x7fffu + lsb;
    return (unsigned short)(u >> 16);
}

// CDNA5 async global->LDS copy (16B per lane), tracked by ASYNCcnt.
// ldsOff: LDS byte offset (low 32 bits of the flat shared-memory address).
// gaddr : 64-bit global byte address.
__device__ __forceinline__ void async_b128(unsigned ldsOff, unsigned long long gaddr) {
    asm volatile("global_load_async_to_lds_b128 %0, %1, off"
                 :: "v"(ldsOff), "v"(gaddr)
                 : "memory");
}

__device__ __forceinline__ void wait_async0() {
    asm volatile("s_wait_asynccnt 0x0" ::: "memory");
}

// ---------------------------------------------------------------------------
// Kernel 1: gather + f32->bf16 pack + per-pair cosine (f32) + pts pack
// One 256-thread block per (b, s) row.
// ---------------------------------------------------------------------------
__global__ __launch_bounds__(256)
void gather_pack_kernel(const float* __restrict__ velo_feat,
                        const float* __restrict__ velo_pts,
                        const float* __restrict__ ref_feat,
                        const int*   __restrict__ matches,
                        unsigned short* __restrict__ Abf,
                        unsigned short* __restrict__ Pbf,
                        float4* __restrict__ aPts,
                        float*  __restrict__ dPos) {
    const int bsrow = blockIdx.x;            // b*SM + s
    const int b   = bsrow >> 11;
    const int tid = threadIdx.x;
    const int ia = matches[(size_t)bsrow * 2 + 0];
    const int ip = matches[(size_t)bsrow * 2 + 1];
    const float* arow = velo_feat + ((size_t)b * NPTS + ia) * FDIM;
    const float* prow = ref_feat  + ((size_t)b * NPTS + ip) * FDIM;
    unsigned short* aout = Abf + (size_t)bsrow * FDIM;
    unsigned short* pout = Pbf + (size_t)bsrow * FDIM;

    float dot = 0.f, na = 0.f, np2 = 0.f;
    for (int i = tid; i < FDIM; i += 256) {
        float a = arow[i], p = prow[i];
        aout[i] = f32_to_bf16_rne(a);
        pout[i] = f32_to_bf16_rne(p);
        dot = fmaf(a, p, dot);
        na  = fmaf(a, a, na);
        np2 = fmaf(p, p, np2);
    }
    #pragma unroll
    for (int m = 16; m >= 1; m >>= 1) {
        dot += __shfl_xor(dot, m);
        na  += __shfl_xor(na,  m);
        np2 += __shfl_xor(np2, m);
    }
    __shared__ float sd[8], sa[8], sp[8];
    const int w = tid >> 5, lane = tid & 31;
    if (lane == 0) { sd[w] = dot; sa[w] = na; sp[w] = np2; }
    __syncthreads();
    if (tid == 0) {
        float D = 0.f, A = 0.f, P = 0.f;
        #pragma unroll
        for (int i = 0; i < 8; ++i) { D += sd[i]; A += sa[i]; P += sp[i]; }
        float dp = D / fmaxf(sqrtf(A) * sqrtf(P), 1e-8f);
        dPos[bsrow] = dp;
        const float* pp = velo_pts + ((size_t)b * NPTS + ia) * 3;
        float x = pp[0], y = pp[1], z = pp[2];
        aPts[bsrow] = make_float4(x, y, z, x * x + y * y + z * z);
    }
}

// ---------------------------------------------------------------------------
// Kernel 2: fused bf16 WMMA GEMM (D = A * P^T) + distance mask + exp + rowsum.
// Block = 256 threads (8 waves). Block tile = 128x128, K chunk = 32.
// Staging: double-buffered CDNA5 async global->LDS (ASYNCcnt), prefetching
// K-step k+1 while computing step k. Buffer flip is a XOR'd offset and the
// K-loop is kept single-body (unroll 1) so accumulators stay pinned in one
// VGPR block (no per-iteration accumulator moves).
// ---------------------------------------------------------------------------
__global__ __launch_bounds__(256)
void nce_gemm_kernel(const unsigned short* __restrict__ Abf,
                     const unsigned short* __restrict__ Pbf,
                     const float4* __restrict__ aPts,
                     float* __restrict__ partial) {
    const int b   = blockIdx.z;
    const int r0  = blockIdx.y * BM;
    const int t0  = blockIdx.x * BN;
    const int tid = threadIdx.x;

    __shared__ unsigned short ldsA[2 * BM * BK];
    __shared__ unsigned short ldsB[2 * BN * BK];
    __shared__ float4 ptsR[BM];
    __shared__ float4 ptsC[BN];

    if (tid < BM) ptsR[tid]      = aPts[(size_t)b * SM + r0 + tid];
    else          ptsC[tid - BM] = aPts[(size_t)b * SM + t0 + (tid - BM)];

    const v8f vzero = {0.f, 0.f, 0.f, 0.f, 0.f, 0.f, 0.f, 0.f};
    v8f acc[8];
    #pragma unroll
    for (int i = 0; i < 8; ++i) acc[i] = vzero;

    const unsigned short* Ag = Abf + ((size_t)b * SM + r0) * FDIM;
    const unsigned short* Bg = Pbf + ((size_t)b * SM + t0) * FDIM;
    const int lrow   = tid >> 1;      // 0..127 : tile row this thread stages
    const int lchunk = tid & 1;       // which 16-half (32B) chunk of the K=32 row
    const int w    = tid >> 5;        // wave id 0..7
    const int lane = tid & 31;
    const int h    = lane >> 4;       // lane half (0/1)
    const int ln   = lane & 15;

    // LDS byte offsets (flat shared addresses carry LDS offset in low 32 bits)
    const unsigned ldsAoff = (unsigned)(uintptr_t)(&ldsA[0]);
    const unsigned ldsBoff = (unsigned)(uintptr_t)(&ldsB[0]);

    // Per-thread staging slot within a buffer (in halfs) and global row base.
    const unsigned slotH = (unsigned)(lrow * BK + lchunk * 16);
    const size_t   gOffH = (size_t)lrow * FDIM + lchunk * 16;

    auto issue_tile = [&](unsigned bufOffH, int k) {
        const unsigned short* ga = Ag + gOffH + k;
        const unsigned short* gb = Bg + gOffH + k;
        unsigned la = ldsAoff + (bufOffH + slotH) * 2u;
        unsigned lb = ldsBoff + (bufOffH + slotH) * 2u;
        async_b128(la,       (unsigned long long)(uintptr_t)ga);
        async_b128(la + 16u, (unsigned long long)(uintptr_t)(ga + 8));
        async_b128(lb,       (unsigned long long)(uintptr_t)gb);
        async_b128(lb + 16u, (unsigned long long)(uintptr_t)(gb + 8));
    };

    issue_tile(0u, 0);   // prologue prefetch

    unsigned curOffH = 0u;                 // current buffer offset in halfs
    const unsigned flipH = (unsigned)(BM * BK);

    #pragma unroll 1
    for (int kt = 0; kt < NKT; ++kt) {
        wait_async0();      // our async writes for the current buffer landed
        __syncthreads();    // everyone's writes visible; prev-iter readers done
        const unsigned nxtOffH = curOffH ^ flipH;
        if (kt + 1 < NKT)
            issue_tile(nxtOffH, (kt + 1) * BK);  // overlaps WMMAs below

        // A fragment: row M = w*16 + ln, K split per lane-half
        V16Frag af;
        const unsigned short* arp = ldsA + curOffH + (w * 16 + ln) * BK;
        af.q[0] = *(const uint4*)(arp + h * 16);
        af.q[1] = *(const uint4*)(arp + h * 16 + 8);

        const unsigned short* brow = ldsB + curOffH + ln * BK + h * 16;
        #pragma unroll
        for (int c = 0; c < 8; ++c) {
            V16Frag bfr;
            const unsigned short* brp = brow + c * (16 * BK);
            bfr.q[0] = ((const uint4*)brp)[0];
            bfr.q[1] = ((const uint4*)brp)[1];
            acc[c] = __builtin_amdgcn_wmma_f32_16x16x32_bf16(
                false, af.bf, false, bfr.bf, (short)0, acc[c], false, false);
        }
        curOffH = nxtOffH;
    }

    // Epilogue: mask by pairwise distance, exp, per-row partial sums.
    float rowSum[8];
    #pragma unroll
    for (int v = 0; v < 8; ++v) rowSum[v] = 0.f;
    #pragma unroll
    for (int c = 0; c < 8; ++c) {
        float4 pc = ptsC[c * 16 + ln];
        #pragma unroll
        for (int v = 0; v < 8; ++v) {
            float4 pr = ptsR[w * 16 + v + 8 * h];   // D row M = v + 8*h
            float d2 = pr.w + pc.w - 2.f * (pr.x * pc.x + pr.y * pc.y + pr.z * pc.z);
            float e  = (d2 > 25.f) ? exp2f(acc[c][v] * K_EXP2) : 0.f;
            rowSum[v] += e;
        }
    }
    #pragma unroll
    for (int v = 0; v < 8; ++v) {
        float s = rowSum[v];
        s += __shfl_xor(s, 1);
        s += __shfl_xor(s, 2);
        s += __shfl_xor(s, 4);
        s += __shfl_xor(s, 8);
        if (ln == 0) {
            int srow = r0 + w * 16 + v + 8 * h;
            partial[((size_t)b * SM + srow) * NCB + blockIdx.x] = s;
        }
    }
}

// ---------------------------------------------------------------------------
// Kernel 3: final deterministic reduction -> scalar mean loss.
// ---------------------------------------------------------------------------
__global__ __launch_bounds__(256)
void finalize_kernel(const float* __restrict__ dPos,
                     const float* __restrict__ partial,
                     float* __restrict__ out) {
    const int tid = threadIdx.x;
    float acc = 0.f;
    for (int i = tid; i < BS * SM; i += 256) {
        float dp = dPos[i];
        float denom = exp2f(dp * K_EXP2);       // positive-pair term (mask == 1)
        const float* p = partial + (size_t)i * NCB;
        #pragma unroll
        for (int cb = 0; cb < NCB; ++cb) denom += p[cb];
        acc += logf(denom) - dp * 10.0f;        // -log(exp(dp/tau)/denom)
    }
    #pragma unroll
    for (int m = 16; m >= 1; m >>= 1) acc += __shfl_xor(acc, m);
    __shared__ float s[8];
    if ((tid & 31) == 0) s[tid >> 5] = acc;
    __syncthreads();
    if (tid == 0) {
        float t = 0.f;
        #pragma unroll
        for (int i = 0; i < 8; ++i) t += s[i];
        out[0] = t / (float)(BS * SM);
    }
}

extern "C" void kernel_launch(void* const* d_in, const int* in_sizes, int n_in,
                              void* d_out, int out_size, void* d_ws, size_t ws_size,
                              hipStream_t stream) {
    (void)in_sizes; (void)n_in; (void)out_size; (void)ws_size;
    const float* velo_feat = (const float*)d_in[0];
    const float* velo_pts  = (const float*)d_in[1];
    const float* ref_feat  = (const float*)d_in[2];
    const int*   matches   = (const int*)d_in[3];

    char* ws = (char*)d_ws;
    size_t off = 0;
    unsigned short* Abf = (unsigned short*)(ws + off); off += (size_t)BS * SM * FDIM * 2;
    unsigned short* Pbf = (unsigned short*)(ws + off); off += (size_t)BS * SM * FDIM * 2;
    float4* aPts        = (float4*)(ws + off);         off += (size_t)BS * SM * sizeof(float4);
    float*  dPos        = (float*)(ws + off);          off += (size_t)BS * SM * sizeof(float);
    float*  partial     = (float*)(ws + off);          off += (size_t)BS * SM * NCB * sizeof(float);

    gather_pack_kernel<<<BS * SM, 256, 0, stream>>>(
        velo_feat, velo_pts, ref_feat, matches, Abf, Pbf, aPts, dPos);

    dim3 g2(SM / BN, SM / BM, BS);
    nce_gemm_kernel<<<g2, 256, 0, stream>>>(Abf, Pbf, aPts, partial);

    finalize_kernel<<<1, 256, 0, stream>>>(dPos, partial, (float*)d_out);
}
